// PerformerLM_77068893159620
// MI455X (gfx1250) — matrix operand order repaired and checked by reading, not compiled, not run on hardware
//
#include <hip/hip_runtime.h>
#include <math.h>

// ---------------------------------------------------------------------------
// PerformerLM forward for MI455X (gfx1250, wave32, WMMA).
// All matmuls go through v_wmma_f32_16x16x32_bf16 (f32 accumulate).
// GEMM: 128x64 block tile, 8 waves, 32x32 per wave (4 WMMAs), double-buffered
// LDS with one barrier per 32-K step, software-pipelined global loads.
// Workspace requirement: ~150 MB (see offsets below).
// ---------------------------------------------------------------------------

typedef __bf16 bf16_t;
typedef bf16_t v16bf __attribute__((ext_vector_type(16)));
typedef float  v8f   __attribute__((ext_vector_type(8)));

#define N_SEQ 2048
#define DIMM  1024
#define HEADS 16
#define DHD   64
#define NBF   256
#define FFH   4096
#define VOCAB 32000
#define DEPTH 4

#define MB (1ull << 20)
// workspace layout (bytes)
#define OFF_X     (0ull  * MB)   // f32  [2048,1024] residual stream      8 MB
#define OFF_QBF   (8ull  * MB)   // bf16 [2048,1024]                      4 MB
#define OFF_KBF   (12ull * MB)   // bf16 [2048,1024]                      4 MB
#define OFF_VBF   (16ull * MB)   // bf16 [2048,1024]                      4 MB
#define OFF_XN    (20ull * MB)   // bf16 [2048,1024] layernorm output     4 MB
#define OFF_ATTN  (24ull * MB)   // bf16 [2048,1024] merged-head attn     4 MB
#define OFF_H1    (28ull * MB)   // bf16 [2048,4096] FFN hidden          16 MB
#define OFF_T     (44ull * MB)   // f32  [2048,1024] pre-residual temp    8 MB
#define OFF_QP    (52ull * MB)   // bf16 [16,2048,256] q features        16 MB
#define OFF_KPT   (68ull * MB)   // bf16 [16,256,2048] k features^T      16 MB
#define OFF_CTX   (84ull * MB)   // bf16 [16,256,64] context              0.5 MB
#define OFF_KSUM  (85ull * MB)   // f32  [16,256]
#define OFF_PROJT (86ull * MB)   // bf16 [4,64,256] transposed projs
#define OFF_WBF   (87ull * MB)   // bf16 weight scratch, up to 62.5 MB
// total ~150 MB

__device__ __forceinline__ unsigned short f2bf(float f) {
  unsigned int u = __builtin_bit_cast(unsigned int, f);
  u += 0x7FFFu + ((u >> 16) & 1u);                    // round-to-nearest-even
  return (unsigned short)(u >> 16);
}
__device__ __forceinline__ float bf2f(unsigned short h) {
  return __builtin_bit_cast(float, ((unsigned int)h) << 16);
}
__device__ __forceinline__ float gelu_tanh(float x) {
  return 0.5f * x * (1.0f + tanhf(0.7978845608028654f * (x + 0.044715f * x * x * x)));
}
__device__ __forceinline__ float wave_sum(float v) {
  #pragma unroll
  for (int off = 16; off > 0; off >>= 1) v += __shfl_down(v, off, 32);
  return v;
}

union FragBF { uint4 u[2]; v16bf v; };

// ---------------------------------------------------------------------------
// Generic batched WMMA GEMM:  C = act(alpha * A[MxK] @ B[KxN] + bias)
// A,B bf16 row-major with leading dims lda/ldb and batch strides sA/sB.
// ACT: 0 = none, 1 = gelu(tanh), 2 = relu(x)+1e-3 (FAVOR+ feature map)
// OBF: 1 = bf16 output, 0 = f32 output.  TRS: 1 = store C transposed.
// Requires M%128==0, N%64==0, K%32==0 (true for every call here).
// Block tile 128x64, 8 waves of 32x32 (4 WMMAs each), LDS double-buffered.
// ---------------------------------------------------------------------------
template <int ACT, int OBF, int TRS>
__global__ __launch_bounds__(256) void wmma_gemm_kernel(
    const unsigned short* __restrict__ A, const unsigned short* __restrict__ B,
    const float* __restrict__ bias, void* __restrict__ Cout,
    int M, int N, int K, int lda, int ldb, int ldc,
    long long sA, long long sB, long long sC, float alpha) {
  constexpr int AP = 40, BP = 40;                     // padded LDS strides
  __shared__ unsigned short Asub[2][128 * AP];        // [buf][128 rows][32 k]
  __shared__ unsigned short BsubT[2][64 * BP];        // [buf][64 cols][32 k]

  const int z = blockIdx.z;
  A += (long long)z * sA;
  B += (long long)z * sB;

  const int m0 = blockIdx.y * 128;
  const int n0 = blockIdx.x * 64;
  const int tid = threadIdx.x;
  const int lane = tid & 31;
  const int wid = tid >> 5;                           // 8 waves
  const int wr = wid >> 1;                            // wave row 0..3 (32 M each)
  const int wc = wid & 1;                             // wave col 0..1 (32 N each)

  // A tile: 128x32 = 4096 elems -> 2 uint4 per thread
  const int a_row = tid >> 1;                         // 0..127
  const int a_col = (tid & 1) << 4;                   // 0 or 16 (chunks +0, +8)
  // B tile: 32x64 = 2048 elems -> 1 uint4 per thread
  const int b_row = tid >> 3;                         // 0..31
  const int b_col = (tid & 7) << 3;                   // 0..56

  const int laneH = lane & 15;
  const int kbA = (lane < 16) ? 0 : 8;                // 16-bit A layout (ISA 7.12.2)
  const int kbB = (lane < 16) ? 0 : 8;                // 16-bit B layout

  v8f acc00 = {}, acc01 = {}, acc10 = {}, acc11 = {};

  const unsigned short* aptr = A + (long long)(m0 + a_row) * lda + a_col;
  const unsigned short* bptr = B + (long long)b_row * ldb + (n0 + b_col);
  const long long bstep = (long long)32 * ldb;

  // ---- prologue: stage tile 0 ----
  {
    uint4 av0 = *(const uint4*)(aptr);
    uint4 av1 = *(const uint4*)(aptr + 8);
    uint4 bv  = *(const uint4*)(bptr);
    *(uint4*)(&Asub[0][a_row * AP + a_col])     = av0;
    *(uint4*)(&Asub[0][a_row * AP + a_col + 8]) = av1;
    const unsigned short* bh = (const unsigned short*)&bv;
    #pragma unroll
    for (int e = 0; e < 8; ++e)                       // transpose on LDS store
      BsubT[0][(b_col + e) * BP + b_row] = bh[e];
  }
  __syncthreads();

  for (int k0 = 0; k0 < K; k0 += 32) {
    const int cur = (k0 >> 5) & 1;
    const int nxt = cur ^ 1;
    const bool has_next = (k0 + 32 < K);

    // software pipeline: issue next-tile global loads before computing
    uint4 na0, na1, nb;
    if (has_next) {
      na0 = *(const uint4*)(aptr + (k0 + 32));
      na1 = *(const uint4*)(aptr + (k0 + 32) + 8);
      nb  = *(const uint4*)(bptr + (long long)(k0 + 32) / 32 * bstep);
      if (k0 + 64 < K) {                              // prime L2 one tile further
        __builtin_prefetch(aptr + (k0 + 64), 0, 1);
        __builtin_prefetch(bptr + (long long)(k0 + 64) / 32 * bstep, 0, 1);
      }
    }

    // fragments from current LDS buffer
    FragBF fa0, fa1, fb0, fb1;
    fa0.u[0] = *(const uint4*)(&Asub[cur][(wr * 32 + laneH) * AP + kbA]);
    fa0.u[1] = *(const uint4*)(&Asub[cur][(wr * 32 + laneH) * AP + 16 + kbA]);
    fa1.u[0] = *(const uint4*)(&Asub[cur][(wr * 32 + 16 + laneH) * AP + kbA]);
    fa1.u[1] = *(const uint4*)(&Asub[cur][(wr * 32 + 16 + laneH) * AP + 16 + kbA]);
    fb0.u[0] = *(const uint4*)(&BsubT[cur][(wc * 32 + laneH) * BP + kbB]);
    fb0.u[1] = *(const uint4*)(&BsubT[cur][(wc * 32 + laneH) * BP + kbB + 16]);
    fb1.u[0] = *(const uint4*)(&BsubT[cur][(wc * 32 + 16 + laneH) * BP + kbB]);
    fb1.u[1] = *(const uint4*)(&BsubT[cur][(wc * 32 + 16 + laneH) * BP + kbB + 16]);

    acc00 = __builtin_amdgcn_wmma_f32_16x16x32_bf16(false, fa0.v, false, fb0.v,
                                                    (short)0, acc00, false, false);
    acc01 = __builtin_amdgcn_wmma_f32_16x16x32_bf16(false, fa0.v, false, fb1.v,
                                                    (short)0, acc01, false, false);
    acc10 = __builtin_amdgcn_wmma_f32_16x16x32_bf16(false, fa1.v, false, fb0.v,
                                                    (short)0, acc10, false, false);
    acc11 = __builtin_amdgcn_wmma_f32_16x16x32_bf16(false, fa1.v, false, fb1.v,
                                                    (short)0, acc11, false, false);

    if (has_next) {                                   // fill the other buffer
      *(uint4*)(&Asub[nxt][a_row * AP + a_col])     = na0;
      *(uint4*)(&Asub[nxt][a_row * AP + a_col + 8]) = na1;
      const unsigned short* bh = (const unsigned short*)&nb;
      #pragma unroll
      for (int e = 0; e < 8; ++e)
        BsubT[nxt][(b_col + e) * BP + b_row] = bh[e];
      __syncthreads();                                // one barrier per K-step
    }
  }

  // C/D layout: VGPR r holds (M = r + 8*(lane>=16), N = lane&15)
  const int rb = m0 + wr * 32 + ((lane >> 4) << 3);
  const int cb = n0 + wc * 32 + laneH;
  const v8f accs[4] = {acc00, acc01, acc10, acc11};
  #pragma unroll
  for (int t = 0; t < 4; ++t) {
    const int mi = t >> 1, ni = t & 1;
    const int rbase = rb + mi * 16;
    const int col = cb + ni * 16;
    const float bs = bias ? bias[col] : 0.0f;
    v8f av = accs[t];
    #pragma unroll
    for (int r = 0; r < 8; ++r) {
      float val = av[r] * alpha + bs;
      if (ACT == 1) val = gelu_tanh(val);
      if (ACT == 2) val = fmaxf(val, 0.0f) + 1e-3f;
      long long idx = TRS ? ((long long)col * ldc + (rbase + r))
                          : ((long long)(rbase + r) * ldc + col);
      idx += (long long)z * sC;
      if (OBF) ((unsigned short*)Cout)[idx] = f2bf(val);
      else     ((float*)Cout)[idx] = val;
    }
  }
}

// ---------------------------------------------------------------------------
// Elementwise / reduction helpers
// ---------------------------------------------------------------------------
__global__ __launch_bounds__(256) void embed_kernel(
    const int* __restrict__ tok, const float* __restrict__ te,
    const float* __restrict__ pe, float* __restrict__ x) {
  const int n = blockIdx.x;
  const int t = tok[n];
  #pragma unroll
  for (int j = 0; j < 4; ++j) {
    const int d = threadIdx.x + 256 * j;
    x[(long long)n * DIMM + d] = te[(long long)t * DIMM + d] + pe[(long long)n * DIMM + d];
  }
}

__global__ __launch_bounds__(256) void layernorm_kernel(
    const float* __restrict__ x, const float* __restrict__ g,
    const float* __restrict__ b, unsigned short* __restrict__ y) {
  const int row = blockIdx.x;
  const float* xr = x + (long long)row * DIMM;
  float v[4];
  float s = 0.0f;
  #pragma unroll
  for (int j = 0; j < 4; ++j) { v[j] = xr[threadIdx.x + 256 * j]; s += v[j]; }
  __shared__ float red[8];
  const int lane = threadIdx.x & 31, wid = threadIdx.x >> 5;
  s = wave_sum(s);
  if (lane == 0) red[wid] = s;
  __syncthreads();
  float tot = 0.0f;
  #pragma unroll
  for (int j = 0; j < 8; ++j) tot += red[j];
  const float mean = tot * (1.0f / DIMM);
  __syncthreads();
  float ss = 0.0f;
  #pragma unroll
  for (int j = 0; j < 4; ++j) { const float d = v[j] - mean; ss += d * d; }
  ss = wave_sum(ss);
  if (lane == 0) red[wid] = ss;
  __syncthreads();
  float tot2 = 0.0f;
  #pragma unroll
  for (int j = 0; j < 8; ++j) tot2 += red[j];
  const float rs = rsqrtf(tot2 * (1.0f / DIMM) + 1e-3f);
  unsigned short* yr = y + (long long)row * DIMM;
  #pragma unroll
  for (int j = 0; j < 4; ++j) {
    const int d = threadIdx.x + 256 * j;
    yr[d] = f2bf((v[j] - mean) * rs * g[d] + b[d]);
  }
}

__global__ __launch_bounds__(256) void cvt_bf16_kernel(
    const float* __restrict__ w, unsigned short* __restrict__ o, long long n) {
  const long long i = ((long long)blockIdx.x * 256 + threadIdx.x) * 4;
  if (i >= n) return;
  const float4 f = *(const float4*)(w + i);
  uint2 p;
  p.x = (unsigned)f2bf(f.x) | ((unsigned)f2bf(f.y) << 16);
  p.y = (unsigned)f2bf(f.z) | ((unsigned)f2bf(f.w) << 16);
  *(uint2*)(o + i) = p;
}

// projT[l][d][m] = proj[l][m][d]  (bf16), so feature-map GEMM B is row-major
__global__ __launch_bounds__(256) void projT_kernel(
    const float* __restrict__ proj, unsigned short* __restrict__ pt) {
  const int idx = blockIdx.x * 256 + threadIdx.x;   // DEPTH*NBF*DHD = 65536
  const int l = idx >> 14;
  const int r = idx & 16383;
  const int m = r >> 6, d = r & 63;
  pt[(l * DHD + d) * NBF + m] = f2bf(proj[idx]);
}

// ksum[h][m] = sum_n kpT[h][m][n]; one wave per (h,m) row of 2048
__global__ __launch_bounds__(256) void ksum_kernel(
    const unsigned short* __restrict__ kpT, float* __restrict__ ksum) {
  const int row = blockIdx.x * 8 + (threadIdx.x >> 5);   // 0..HEADS*NBF-1
  const int lane = threadIdx.x & 31;
  const unsigned short* p = kpT + (long long)row * N_SEQ;
  float s = 0.0f;
  #pragma unroll
  for (int j = 0; j < 8; ++j) {
    const uint4 u = *(const uint4*)(p + lane * 8 + j * 256);
    const unsigned short* h = (const unsigned short*)&u;
    #pragma unroll
    for (int e = 0; e < 8; ++e) s += bf2f(h[e]);
  }
  s = wave_sum(s);
  if (lane == 0) ksum[row] = s;
}

// dinv[h][n] = 1/(qp[h][n][:] . ksum[h][:]); fold into qp by row-rescale
__global__ __launch_bounds__(256) void dinv_scale_kernel(
    unsigned short* __restrict__ qp, const float* __restrict__ ksum) {
  const int row = blockIdx.x * 8 + (threadIdx.x >> 5);   // h*2048 + n
  const int lane = threadIdx.x & 31;
  const int h = row >> 11;
  unsigned short* p = qp + (long long)row * NBF;
  const float* ks = ksum + h * NBF;
  uint4 u = *(const uint4*)(p + lane * 8);
  const unsigned short* hh = (const unsigned short*)&u;
  float s = 0.0f;
  #pragma unroll
  for (int e = 0; e < 8; ++e) s += bf2f(hh[e]) * ks[lane * 8 + e];
  s = wave_sum(s);
  s = __shfl(s, 0, 32);
  const float d = 1.0f / s;
  uint4 o;
  unsigned short* oh = (unsigned short*)&o;
  #pragma unroll
  for (int e = 0; e < 8; ++e) oh[e] = f2bf(bf2f(hh[e]) * d);
  *(uint4*)(p + lane * 8) = o;
}

__global__ __launch_bounds__(256) void resid_kernel(
    float* __restrict__ x, const float* __restrict__ t) {
  const long long i = ((long long)blockIdx.x * 256 + threadIdx.x) * 4;
  float4 a = *(const float4*)(x + i);
  const float4 b = *(const float4*)(t + i);
  a.x += b.x; a.y += b.y; a.z += b.z; a.w += b.w;
  *(float4*)(x + i) = a;
}

// ---------------------------------------------------------------------------
extern "C" void kernel_launch(void* const* d_in, const int* in_sizes, int n_in,
                              void* d_out, int out_size, void* d_ws, size_t ws_size,
                              hipStream_t stream) {
  (void)in_sizes; (void)n_in; (void)out_size; (void)ws_size;
  const int*   tokens    = (const int*)d_in[0];
  const float* token_emb = (const float*)d_in[1];
  const float* pos_emb   = (const float*)d_in[2];
  // layers: 16 arrays each, starting at index 3
  auto LW = [&](int l, int j) { return (const float*)d_in[3 + 16 * l + j]; };
  const float* projs   = (const float*)d_in[67];
  const float* norm_g  = (const float*)d_in[68];
  const float* norm_b  = (const float*)d_in[69];
  const float* dense_w = (const float*)d_in[70];
  const float* dense_b = (const float*)d_in[71];

  char* ws = (char*)d_ws;
  float*          x     = (float*)(ws + OFF_X);
  unsigned short* qbf   = (unsigned short*)(ws + OFF_QBF);
  unsigned short* kbf   = (unsigned short*)(ws + OFF_KBF);
  unsigned short* vbf   = (unsigned short*)(ws + OFF_VBF);
  unsigned short* xn    = (unsigned short*)(ws + OFF_XN);
  unsigned short* attn  = (unsigned short*)(ws + OFF_ATTN);
  unsigned short* h1    = (unsigned short*)(ws + OFF_H1);
  float*          tbuf  = (float*)(ws + OFF_T);
  unsigned short* qp    = (unsigned short*)(ws + OFF_QP);
  unsigned short* kpT   = (unsigned short*)(ws + OFF_KPT);
  unsigned short* ctx   = (unsigned short*)(ws + OFF_CTX);
  float*          ksum  = (float*)(ws + OFF_KSUM);
  unsigned short* projT = (unsigned short*)(ws + OFF_PROJT);
  unsigned short* wbf   = (unsigned short*)(ws + OFF_WBF);

  const float DN = 0.3535533905932738f;  // 64^-0.25

  embed_kernel<<<N_SEQ, 256, 0, stream>>>(tokens, token_emb, pos_emb, x);
  projT_kernel<<<(DEPTH * NBF * DHD) / 256, 256, 0, stream>>>(projs, projT);

  for (int l = 0; l < DEPTH; ++l) {
    const float *ln1g = LW(l,0), *ln1b = LW(l,1), *wq = LW(l,2), *bq = LW(l,3),
                *wk = LW(l,4), *bk = LW(l,5), *wv = LW(l,6), *bv = LW(l,7),
                *wo = LW(l,8), *bo = LW(l,9), *ln2g = LW(l,10), *ln2b = LW(l,11),
                *w1 = LW(l,12), *b1 = LW(l,13), *w2 = LW(l,14), *b2 = LW(l,15);

    // --- attention block (prenorm) ---
    layernorm_kernel<<<N_SEQ, 256, 0, stream>>>(x, ln1g, ln1b, xn);

    const long long DD = (long long)DIMM * DIMM;
    cvt_bf16_kernel<<<DD / 1024, 256, 0, stream>>>(wq, wbf, DD);
    wmma_gemm_kernel<0,1,0><<<dim3(DIMM/64, N_SEQ/128, 1), 256, 0, stream>>>(
        xn, wbf, bq, qbf, N_SEQ, DIMM, DIMM, DIMM, DIMM, DIMM, 0, 0, 0, 1.0f);
    cvt_bf16_kernel<<<DD / 1024, 256, 0, stream>>>(wk, wbf, DD);
    wmma_gemm_kernel<0,1,0><<<dim3(DIMM/64, N_SEQ/128, 1), 256, 0, stream>>>(
        xn, wbf, bk, kbf, N_SEQ, DIMM, DIMM, DIMM, DIMM, DIMM, 0, 0, 0, 1.0f);
    cvt_bf16_kernel<<<DD / 1024, 256, 0, stream>>>(wv, wbf, DD);
    wmma_gemm_kernel<0,1,0><<<dim3(DIMM/64, N_SEQ/128, 1), 256, 0, stream>>>(
        xn, wbf, bv, vbf, N_SEQ, DIMM, DIMM, DIMM, DIMM, DIMM, 0, 0, 0, 1.0f);

    // qp[h][n][m] = relu(DN * q_h @ projT) + eps   (batched over heads)
    wmma_gemm_kernel<2,1,0><<<dim3(NBF/64, N_SEQ/128, HEADS), 256, 0, stream>>>(
        qbf, projT + (long long)l * DHD * NBF, nullptr, qp,
        N_SEQ, NBF, DHD, DIMM, NBF, NBF,
        (long long)DHD, 0, (long long)N_SEQ * NBF, DN);
    // kpT[h][m][n] (transposed store)
    wmma_gemm_kernel<2,1,1><<<dim3(NBF/64, N_SEQ/128, HEADS), 256, 0, stream>>>(
        kbf, projT + (long long)l * DHD * NBF, nullptr, kpT,
        N_SEQ, NBF, DHD, DIMM, NBF, N_SEQ,
        (long long)DHD, 0, (long long)NBF * N_SEQ, DN);

    ksum_kernel<<<(HEADS * NBF) / 8, 256, 0, stream>>>(kpT, ksum);
    dinv_scale_kernel<<<(HEADS * N_SEQ) / 8, 256, 0, stream>>>(qp, ksum);

    // ctx[h][m][d] = kpT_h @ v_h   (K = 2048)
    wmma_gemm_kernel<0,1,0><<<dim3(DHD/64, NBF/128, HEADS), 256, 0, stream>>>(
        kpT, vbf, nullptr, ctx, NBF, DHD, N_SEQ, N_SEQ, DIMM, DHD,
        (long long)NBF * N_SEQ, (long long)DHD, (long long)NBF * DHD, 1.0f);
    // attn[n][h*64+d] = (dinv-scaled qp_h) @ ctx_h
    wmma_gemm_kernel<0,1,0><<<dim3(DHD/64, N_SEQ/128, HEADS), 256, 0, stream>>>(
        qp, ctx, nullptr, attn, N_SEQ, DHD, NBF, NBF, DHD, DIMM,
        (long long)N_SEQ * NBF, (long long)NBF * DHD, (long long)DHD, 1.0f);

    // output projection + residual
    cvt_bf16_kernel<<<DD / 1024, 256, 0, stream>>>(wo, wbf, DD);
    wmma_gemm_kernel<0,0,0><<<dim3(DIMM/64, N_SEQ/128, 1), 256, 0, stream>>>(
        attn, wbf, bo, tbuf, N_SEQ, DIMM, DIMM, DIMM, DIMM, DIMM, 0, 0, 0, 1.0f);
    resid_kernel<<<(N_SEQ * DIMM) / 1024, 256, 0, stream>>>(x, tbuf);

    // --- FFN block (prenorm) ---
    layernorm_kernel<<<N_SEQ, 256, 0, stream>>>(x, ln2g, ln2b, xn);
    const long long DF = (long long)DIMM * FFH;
    cvt_bf16_kernel<<<DF / 1024, 256, 0, stream>>>(w1, wbf, DF);
    wmma_gemm_kernel<1,1,0><<<dim3(FFH/64, N_SEQ/128, 1), 256, 0, stream>>>(
        xn, wbf, b1, h1, N_SEQ, FFH, DIMM, DIMM, FFH, FFH, 0, 0, 0, 1.0f);
    cvt_bf16_kernel<<<DF / 1024, 256, 0, stream>>>(w2, wbf, DF);
    wmma_gemm_kernel<0,0,0><<<dim3(DIMM/64, N_SEQ/128, 1), 256, 0, stream>>>(
        h1, wbf, b2, tbuf, N_SEQ, DIMM, FFH, FFH, DIMM, DIMM, 0, 0, 0, 1.0f);
    resid_kernel<<<(N_SEQ * DIMM) / 1024, 256, 0, stream>>>(x, tbuf);
  }

  // final layernorm + vocab projection (the big one: 2048x1024x32000)
  layernorm_kernel<<<N_SEQ, 256, 0, stream>>>(x, norm_g, norm_b, xn);
  const long long DV = (long long)DIMM * VOCAB;
  cvt_bf16_kernel<<<DV / 1024, 256, 0, stream>>>(dense_w, wbf, DV);
  wmma_gemm_kernel<0,0,0><<<dim3(VOCAB/64, N_SEQ/128, 1), 256, 0, stream>>>(
      xn, wbf, dense_b, (float*)d_out, N_SEQ, VOCAB, DIMM, DIMM, VOCAB, VOCAB,
      0, 0, 0, 1.0f);
}
